// ModuleNet_8847632630307
// MI455X (gfx1250) — matrix operand order repaired and checked by reading, not compiled, not run on hardware
//
#include <hip/hip_runtime.h>
#include <hip/hip_bf16.h>

typedef __attribute__((ext_vector_type(16))) _Float16     v16h;
typedef __attribute__((ext_vector_type(8)))  _Float16     v8h;
typedef __attribute__((ext_vector_type(8)))  float        v8f;
typedef __attribute__((ext_vector_type(8)))  unsigned int v8u;

#define B_SZ       4096
#define NSTEPS     4
#define E_DIM      128
#define NUM_MOD    64
#define OUT_DIM    16
#define MAX_TILES  320            // sum ceil(cnt_m/16) <= 4096/16 + 64
#define PADB       (MAX_TILES*16) // padded permutation length (5120)
#define LDS_STRIDE 136            // 128 + 8 f16 pad -> no LDS bank conflicts

// ---------------- preprocessing ----------------

__global__ void k_convert_w(const float* __restrict__ W, _Float16* __restrict__ Wh) {
    int i = blockIdx.x * blockDim.x + threadIdx.x;   // exactly 64*128*128 threads
    Wh[i] = (_Float16)W[i];
}

__global__ void k_gather(const int* __restrict__ author_ids,
                         const float* __restrict__ embed,
                         _Float16* __restrict__ X0) {
    int i = blockIdx.x * blockDim.x + threadIdx.x;   // exactly B*E threads
    int s = i >> 7, k = i & 127;
    X0[i] = (_Float16)embed[(size_t)author_ids[s] * E_DIM + k];
}

// ---------------- per-step bucketing ----------------

__global__ void k_zero_counts(int* counts) { counts[threadIdx.x] = 0; }

__global__ void k_hist(const int* __restrict__ module_ids, int step, int* counts) {
    int i = blockIdx.x * blockDim.x + threadIdx.x;
    if (i < B_SZ) atomicAdd(&counts[module_ids[i * NSTEPS + step]], 1);
}

__global__ void k_plan(const int* __restrict__ counts, int* cursor,
                       int* tileModule, int* tileStart, int* nTiles, int* perm) {
    for (int i = threadIdx.x; i < PADB; i += blockDim.x) perm[i] = -1;
    if (threadIdx.x == 0) {
        int ofs = 0, t = 0;
        for (int m = 0; m < NUM_MOD; ++m) {
            cursor[m] = ofs;
            int nt = (counts[m] + 15) >> 4;
            for (int j = 0; j < nt; ++j) { tileModule[t] = m; tileStart[t] = ofs + 16 * j; ++t; }
            ofs += nt * 16;
        }
        *nTiles = t;
    }
}

__global__ void k_scatter(const int* __restrict__ module_ids, int step,
                          int* cursor, int* perm) {
    int i = blockIdx.x * blockDim.x + threadIdx.x;
    if (i < B_SZ) {
        int m = module_ids[i * NSTEPS + step];
        perm[atomicAdd(&cursor[m], 1)] = i;
    }
}

// ---------------- WMMA GEMM step: Y = W_m * X_tile (+bias, [tanh]) ----------------
// 256 threads = 8 waves; wave w computes output rows [16w, 16w+16) for 16 samples.
// Branchless: pad samples read row 0 (garbage stays in their own GEMM column)
// and write to dump row B_SZ.  X tile staged with async global->LDS copies.

template <int LAST>
__global__ void __launch_bounds__(256)
k_gemm_step(const _Float16* __restrict__ Wh, const float* __restrict__ bias,
            const _Float16* __restrict__ Xin, _Float16* __restrict__ Xout,
            float* __restrict__ Xfinal,
            const int* __restrict__ tileModule, const int* __restrict__ tileStart,
            const int* __restrict__ nTiles, const int* __restrict__ perm)
{
    __shared__ _Float16 Xs[16 * LDS_STRIDE];

    if (blockIdx.x >= *nTiles) return;               // uniform across block
    const int mod    = tileModule[blockIdx.x];
    const int tstart = tileStart[blockIdx.x];
    const int tid    = threadIdx.x;

    // ---- async stage 16x128 f16 X tile into LDS: 16 B per thread ----
    {
        int row = tid >> 4;                          // sample slot 0..15
        int col = (tid & 15) * 8;                    // 8 f16 = 16 bytes
        int smp = perm[tstart + row];
        int srcRow = (smp < 0) ? 0 : smp;            // pad -> harmless read of row 0
        unsigned ldsdst = (unsigned)(size_t)&Xs[row * LDS_STRIDE + col];
        const _Float16* src = Xin + (size_t)srcRow * E_DIM + col;
        asm volatile("global_load_async_to_lds_b128 %0, %1, off"
                     :: "v"(ldsdst), "v"(src) : "memory");
        asm volatile("s_wait_asynccnt 0x0" ::: "memory");
    }
    __syncthreads();

    const int w    = tid >> 5;       // wave id: output row tile
    const int lane = tid & 31;
    const int n    = lane & 15;      // A: row M / B: col N / D: col N
    const int kh   = lane >> 4;      // K-half selector
    const int o_base = 16 * w;

    const unsigned int* wrow = reinterpret_cast<const unsigned int*>(
        Wh + ((size_t)mod * E_DIM + (o_base + n)) * E_DIM);
    const unsigned int* xrow = reinterpret_cast<const unsigned int*>(&Xs[n * LDS_STRIDE]);

    v8f c = {0.f, 0.f, 0.f, 0.f, 0.f, 0.f, 0.f, 0.f};
#pragma unroll
    for (int s = 0; s < 4; ++s) {    // K = 128 in 4 slabs of 32
        v8u au, bu;
        // A (16x32 f16): VGPR j<4 -> K = 32s + 8*kh + 2j ; j>=4 -> K = 32s + 16 + 8*kh + 2(j-4)
        const int ab = 16 * s + 4 * kh;
        au[0] = wrow[ab + 0]; au[1] = wrow[ab + 1]; au[2] = wrow[ab + 2]; au[3] = wrow[ab + 3];
        au[4] = wrow[ab + 8]; au[5] = wrow[ab + 9]; au[6] = wrow[ab + 10]; au[7] = wrow[ab + 11];
        // B (32x16 f16): lane n = column (sample), VGPR j<4 -> K = 32s + 8*kh + 2j ; j>=4 -> K = 32s + 16 + 8*kh + 2(j-4)
        const int bb = 16 * s + 4 * kh;
#pragma unroll
        for (int j = 0; j < 8; ++j) bu[j] = xrow[bb + j + (j >= 4 ? 4 : 0)];
        v16h a = __builtin_bit_cast(v16h, au);
        v16h b = __builtin_bit_cast(v16h, bu);
        c = __builtin_amdgcn_wmma_f32_16x16x32_f16(false, a, false, b, (short)0, c, false, false);
    }

    // ---- epilogue: lane owns 8 consecutive channels o = o_base+8*kh .. +7 ----
    const int o0 = o_base + 8 * kh;
    const float* brow = bias + mod * E_DIM + o0;
    const float4 b0 = *reinterpret_cast<const float4*>(brow);
    const float4 b1 = *reinterpret_cast<const float4*>(brow + 4);
    float y[8];
    y[0] = c[0] + b0.x; y[1] = c[1] + b0.y; y[2] = c[2] + b0.z; y[3] = c[3] + b0.w;
    y[4] = c[4] + b1.x; y[5] = c[5] + b1.y; y[6] = c[6] + b1.z; y[7] = c[7] + b1.w;

    const int smp = perm[tstart + n];
    const size_t srow = (smp < 0) ? (size_t)B_SZ : (size_t)smp;   // dump row for pads
    if (LAST) {
        v8f v;
#pragma unroll
        for (int r = 0; r < 8; ++r) v[r] = y[r];
        *reinterpret_cast<v8f*>(Xfinal + srow * E_DIM + o0) = v;
    } else {
        v8h h;
#pragma unroll
        for (int r = 0; r < 8; ++r) h[r] = (_Float16)tanhf(y[r]);
        *reinterpret_cast<v8h*>(Xout + srow * E_DIM + o0) = h;
    }
}

// ---------------- classifier: out = Xf @ cls_W^T + cls_b (f32, tiny) ----------------

__global__ void k_classifier(const float* __restrict__ Xf, const float* __restrict__ cls_W,
                             const float* __restrict__ cls_b, float* __restrict__ out) {
    int i = blockIdx.x * blockDim.x + threadIdx.x;   // exactly B*OUT_DIM threads
    int s = i >> 4, od = i & 15;
    const float4* x4 = reinterpret_cast<const float4*>(Xf + (size_t)s * E_DIM);
    const float4* w4 = reinterpret_cast<const float4*>(cls_W + od * E_DIM);
    float acc = cls_b[od];
#pragma unroll 8
    for (int k = 0; k < E_DIM / 4; ++k) {
        float4 a = x4[k], b = w4[k];
        acc += a.x * b.x + a.y * b.y + a.z * b.z + a.w * b.w;
    }
    out[i] = acc;
}

// ---------------- launch ----------------

extern "C" void kernel_launch(void* const* d_in, const int* in_sizes, int n_in,
                              void* d_out, int out_size, void* d_ws, size_t ws_size,
                              hipStream_t stream) {
    const int*   author_ids = (const int*)  d_in[0];
    const int*   module_ids = (const int*)  d_in[1];
    const float* embed      = (const float*)d_in[2];
    const float* W          = (const float*)d_in[3];
    const float* bias       = (const float*)d_in[4];
    const float* cls_W      = (const float*)d_in[5];
    const float* cls_b      = (const float*)d_in[6];

    // Workspace layout (~6.4 MiB); X buffers have one extra dump row (index B_SZ)
    constexpr size_t WH_OFF = 0;                                   // 2 MiB
    constexpr size_t XA_OFF = WH_OFF + (size_t)(2u << 20);
    constexpr size_t XSZ    = (size_t)(B_SZ + 1) * E_DIM * 2;      // f16, +dump row
    constexpr size_t XB_OFF = XA_OFF + ((XSZ + 4095) & ~4095ull);
    constexpr size_t XF_OFF = XB_OFF + ((XSZ + 4095) & ~4095ull);
    constexpr size_t XFSZ   = (size_t)(B_SZ + 1) * E_DIM * 4;      // f32, +dump row
    constexpr size_t MT_OFF = XF_OFF + ((XFSZ + 4095) & ~4095ull);

    char* ws = (char*)d_ws;
    _Float16* Wh = (_Float16*)(ws + WH_OFF);
    _Float16* Xa = (_Float16*)(ws + XA_OFF);
    _Float16* Xb = (_Float16*)(ws + XB_OFF);
    float*    Xf = (float*)   (ws + XF_OFF);
    char* meta = ws + MT_OFF;
    int* counts     = (int*)(meta);
    int* cursor     = (int*)(meta + 256);
    int* nTiles     = (int*)(meta + 512);
    int* tileModule = (int*)(meta + 1024);
    int* tileStart  = (int*)(meta + 4096);
    int* perm       = (int*)(meta + 8192);                         // PADB ints

    k_convert_w<<<(NUM_MOD * E_DIM * E_DIM) / 256, 256, 0, stream>>>(W, Wh);
    k_gather   <<<(B_SZ * E_DIM) / 256,          256, 0, stream>>>(author_ids, embed, Xa);

    for (int step = 0; step < NSTEPS; ++step) {
        const _Float16* Xin  = (step & 1) ? Xb : Xa;
        _Float16*       Xout = (step & 1) ? Xa : Xb;

        k_zero_counts<<<1, NUM_MOD, 0, stream>>>(counts);
        k_hist   <<<B_SZ / 256, 256, 0, stream>>>(module_ids, step, counts);
        k_plan   <<<1, 256, 0, stream>>>(counts, cursor, tileModule, tileStart, nTiles, perm);
        k_scatter<<<B_SZ / 256, 256, 0, stream>>>(module_ids, step, cursor, perm);
        if (step == NSTEPS - 1)
            k_gemm_step<1><<<MAX_TILES, 256, 0, stream>>>(Wh, bias, Xin, Xout, Xf,
                                                          tileModule, tileStart, nTiles, perm);
        else
            k_gemm_step<0><<<MAX_TILES, 256, 0, stream>>>(Wh, bias, Xin, Xout, Xf,
                                                          tileModule, tileStart, nTiles, perm);
    }

    k_classifier<<<(B_SZ * OUT_DIM) / 256, 256, 0, stream>>>(Xf, cls_W, cls_b, (float*)d_out);
}
